// DifferentialWindowAttention_38362647888579
// MI455X (gfx1250) — matrix operand order repaired and checked by reading, not compiled, not run on hardware
//
#include <hip/hip_runtime.h>
#include <hip/hip_bf16.h>

// ---------------------------------------------------------------------------
// DifferentialWindowAttention for MI455X (gfx1250, wave32, WMMA)
//
// Pipeline (all on `stream`):
//   1) convert x + weights f32 -> bf16 into workspace
//   2) fuse dp_w/proj_w into Wcat (1280x1024) and bias_cat
//   3) GEMM  qkv   = xb @ Wqkv + qkv_b          (bf16 WMMA, f32 accum)
//   4) GEMM  lite  = xb @ [ql|kl|vl]            (bf16 WMMA)
//   5) fused dual-softmax attention per (window, head) -> xcat=[x_out|Δv]
//   6) GEMM  out   = xcat @ Wcat + bias_cat     (f32 output -> d_out)
// ---------------------------------------------------------------------------

typedef __attribute__((ext_vector_type(16))) __bf16   v16bf;
typedef __attribute__((ext_vector_type(8)))  float    v8f;
typedef __attribute__((ext_vector_type(8)))  __bf16   bf16x8;
typedef __attribute__((ext_vector_type(4)))  int      v4i;
typedef __attribute__((ext_vector_type(4)))  unsigned v4u;
typedef __attribute__((ext_vector_type(8)))  unsigned v8u;

static constexpr int kB    = 1024;
static constexpr int kN    = 49;
static constexpr int kC    = 1024;
static constexpr int kH    = 32;
static constexpr int kHD   = 32;
static constexpr int kHDL  = 8;
static constexpr int kRows = kB * kN;                     // 50176
static constexpr float kScale  = 0.17677669529663687f;    // 32^-0.5
static constexpr float kScaleL = 0.35355339059327373f;    // 8^-0.5

#ifndef __has_builtin
#define __has_builtin(x) 0
#endif
#if __has_builtin(__builtin_amdgcn_global_load_async_to_lds_b128)
#define HAVE_ASYNC_LDS 1
#else
#define HAVE_ASYNC_LDS 0
#endif

// ---- bf16 helpers (bit-exact RNE, no reliance on native __bf16 conversions)
static __device__ __forceinline__ __bf16 f2bf(float f) {
  unsigned u = __builtin_bit_cast(unsigned, f);
  unsigned r = u + 0x7FFFu + ((u >> 16) & 1u);
  unsigned short s = (unsigned short)(r >> 16);
  return __builtin_bit_cast(__bf16, s);
}
static __device__ __forceinline__ float bf2f(__bf16 b) {
  unsigned short s = __builtin_bit_cast(unsigned short, b);
  unsigned u = ((unsigned)s) << 16;
  return __builtin_bit_cast(float, u);
}

static __device__ __forceinline__ void store_out(__bf16* p, float v) { *p = f2bf(v); }
static __device__ __forceinline__ void store_out(float*  p, float v) { *p = v; }

// ---- 16B global -> LDS copy: async DMA (ASYNCcnt path) when available.
static __device__ __forceinline__ void copy16_g2l(const void* g, void* l) {
#if HAVE_ASYNC_LDS
  __builtin_amdgcn_global_load_async_to_lds_b128(
      (__attribute__((address_space(1))) v4i*)(unsigned long long)(uintptr_t)g,
      (__attribute__((address_space(3))) v4i*)l, 0, 0);
#else
  *(bf16x8*)l = *(const bf16x8*)g;
#endif
}
static __device__ __forceinline__ void wait_async() {
#if HAVE_ASYNC_LDS
#if __has_builtin(__builtin_amdgcn_s_wait_asynccnt)
  __builtin_amdgcn_s_wait_asynccnt(0);
#else
  asm volatile("s_wait_asynccnt 0x0" ::: "memory");
#endif
#endif
}

// ---------------------------------------------------------------------------
// f32 -> bf16 submatrix convert: dst[r*ld + coff + c] = bf16(src[r*cols + c])
// ---------------------------------------------------------------------------
__global__ void cvt_sub(const float* __restrict__ src, __bf16* __restrict__ dst,
                        long long n, int cols, int ld, int coff) {
  long long i = (long long)blockIdx.x * blockDim.x + threadIdx.x;
  if (i >= n) return;
  long long r = i / cols;
  int c = (int)(i - r * cols);
  dst[r * (long long)ld + coff + c] = f2bf(src[i]);
}

// ---------------------------------------------------------------------------
// Wcat (1280x1024 bf16): rows [0,1024) = proj_w ; rows [1024,1280) =
//   gamma * (dp_w @ proj_w)
// ---------------------------------------------------------------------------
__global__ void fuse_wcat(const float* __restrict__ proj_w,
                          const float* __restrict__ dp_w,
                          const float* __restrict__ gamma,
                          __bf16* __restrict__ wcat) {
  long long i = (long long)blockIdx.x * blockDim.x + threadIdx.x;
  if (i >= (long long)1280 * 1024) return;
  int r = (int)(i >> 10), c = (int)(i & 1023);
  float v;
  if (r < 1024) {
    v = proj_w[(long long)r * 1024 + c];
  } else {
    int k = r - 1024;
    float s = 0.0f;
    for (int j = 0; j < 1024; ++j)
      s += dp_w[(long long)k * 1024 + j] * proj_w[(long long)j * 1024 + c];
    v = gamma[0] * s;
  }
  wcat[i] = f2bf(v);
}

__global__ void fuse_bias(const float* __restrict__ proj_b,
                          const float* __restrict__ dp_b,
                          const float* __restrict__ proj_w,
                          const float* __restrict__ gamma,
                          float* __restrict__ bias_cat) {
  int c = blockIdx.x * blockDim.x + threadIdx.x;
  if (c >= 1024) return;
  float s = 0.0f;
  for (int j = 0; j < 1024; ++j) s += dp_b[j] * proj_w[(long long)j * 1024 + c];
  bias_cat[c] = proj_b[c] + gamma[0] * s;
}

// ---------------------------------------------------------------------------
// bf16 WMMA GEMM: C(MxN) = A(MxK) @ B(KxN) (+ bias[N]), row-major.
// Block 128x128, K-step 32, 8 wave32, 64x32 region per wave (4x2 WMMA tiles).
// A tile staged as straight copy (async DMA when available); B tile staged
// TRANSPOSED into a u32 LDS image ([n][kpair]) so both A and B fragments are
// 2x ds_load_b128 per lane. Double-buffered LDS, one barrier per K-step.
// Requires M%128==N%128==K%32==0 (true for all call sites).
// ---------------------------------------------------------------------------
template <typename OutT>
__global__ __launch_bounds__(256)
void gemm_bf16_wmma(const __bf16* __restrict__ A, const __bf16* __restrict__ B,
                    const float* __restrict__ bias, OutT* __restrict__ C,
                    int M, int N, int K) {
  __shared__ __bf16   As[2][128 * 32];   // [m][k]
  __shared__ unsigned Bt[2][128 * 16];   // [n][kpair]: word j = bf16{k=2j, k=2j+1}

  const int tid  = threadIdx.x;
  const int lane = tid & 31;
  const int wid  = tid >> 5;
  const int wm   = wid >> 2;   // 0..1
  const int wn   = wid & 3;    // 0..3
  const int half = lane >> 4;
  const int lr   = lane & 15;

  const int m0 = blockIdx.y * 128;
  const int n0 = blockIdx.x * 128;

  // A staging: 2 threads per row, 16 bf16 each
  const int arow = tid >> 1, acol = (tid & 1) * 16;
  // B staging: thread owns 2 k-rows x 8 n-cols
  const int bk = (tid >> 4) * 2;    // 0,2,...,30
  const int bn = (tid & 15) * 8;    // 0,8,...,120

  const __bf16* Ag = A + (size_t)(m0 + arow) * K + acol;  // + k0
  const __bf16* Bg = B + (size_t)bk * N + n0 + bn;        // + k0*N

  v8f acc[4][2] = {};

  auto stage = [&](int k0, int buf) {
    // A: straight 2x16B copy (global_load_async_to_lds_b128 when available)
    copy16_g2l(Ag + k0,     &As[buf][arow * 32 + acol]);
    copy16_g2l(Ag + k0 + 8, &As[buf][arow * 32 + acol + 8]);
    // B: transpose while staging. Rows loaded as u32x4 (b128); each word
    // holds two n-adjacent bf16. Repack to K-pairs with integer ops only.
    const v4u q0 = *(const v4u*)(Bg + (size_t)k0 * N);      // row k0+bk
    const v4u q1 = *(const v4u*)(Bg + (size_t)k0 * N + N);  // row k0+bk+1
#pragma unroll
    for (int w = 0; w < 4; ++w) {
      const unsigned even = (q0[w] & 0xFFFFu) | (q1[w] << 16);          // n=bn+2w
      const unsigned odd  = (q0[w] >> 16) | (q1[w] & 0xFFFF0000u);      // n=bn+2w+1
      Bt[buf][(bn + 2 * w) * 16 + (bk >> 1)]     = even;
      Bt[buf][(bn + 2 * w + 1) * 16 + (bk >> 1)] = odd;
    }
  };

  stage(0, 0);
  wait_async();
  __syncthreads();

  int cur = 0;
  for (int k0 = 0; k0 < K; k0 += 32) {
    const int nxt = cur ^ 1;
    if (k0 + 32 < K) stage(k0 + 32, nxt);
    if (k0 + 64 < K) {   // global_prefetch_b8 two tiles ahead
      __builtin_prefetch(Ag + k0 + 64, 0, 1);
      __builtin_prefetch(Bg + (size_t)(k0 + 64) * N, 0, 1);
    }

    // ---- B fragments: per lane N fixed, K contiguous -> 2x ds_load_b128
    v16bf bfrag[2];
#pragma unroll
    for (int nt = 0; nt < 2; ++nt) {
      const unsigned* bq = &Bt[cur][(wn * 32 + nt * 16 + lr) * 16 + 8 * half];
      const v4u w0 = *(const v4u*)bq;        // K = 16*half+0 .. +7
      const v4u w1 = *(const v4u*)(bq + 4);  // K = 16*half+8 .. +15
      const v8u comb = __builtin_shufflevector(w0, w1, 0, 1, 2, 3, 4, 5, 6, 7);
      bfrag[nt] = __builtin_bit_cast(v16bf, comb);
    }
    // ---- A fragments + WMMA (A: two 8-elem K runs per lane)
#pragma unroll
    for (int mt = 0; mt < 4; ++mt) {
      const __bf16* ap = &As[cur][(wm * 64 + mt * 16 + lr) * 32 + 16 * half];
      const bf16x8 a0 = *(const bf16x8*)ap;         // K = 16*half .. +7
      const bf16x8 a1 = *(const bf16x8*)(ap + 8);   // K = 16*half+8 .. +15
      const v16bf af = __builtin_shufflevector(a0, a1, 0, 1, 2, 3, 4, 5, 6, 7,
                                               8, 9, 10, 11, 12, 13, 14, 15);
      acc[mt][0] = __builtin_amdgcn_wmma_f32_16x16x32_bf16(
          false, af, false, bfrag[0], (short)0, acc[mt][0], false, false);
      acc[mt][1] = __builtin_amdgcn_wmma_f32_16x16x32_bf16(
          false, af, false, bfrag[1], (short)0, acc[mt][1], false, false);
    }

    wait_async();        // our async A loads (into nxt) have landed
    __syncthreads();     // everyone done reading cur + all stores visible
    cur = nxt;
  }

  // ---- epilogue: C/D layout — VGPR i: M = i + 8*half, N = lane&15
#pragma unroll
  for (int mt = 0; mt < 4; ++mt) {
#pragma unroll
    for (int nt = 0; nt < 2; ++nt) {
      const int col = n0 + wn * 32 + nt * 16 + lr;
      const float bv = bias ? bias[col] : 0.0f;
#pragma unroll
      for (int i = 0; i < 8; ++i) {
        const int row = m0 + wm * 64 + mt * 16 + half * 8 + i;
        store_out(&C[(size_t)row * N + col], acc[mt][nt][i] + bv);
      }
    }
  }
}

// ---------------------------------------------------------------------------
// Fused dual-branch attention. One block per (window b, head h): 64 threads,
// lane r < 49 owns query row r. Both softmaxes fused so attn_soft (315 MB)
// never hits HBM. Writes xcat row = [x_out | Δv].
// ---------------------------------------------------------------------------
__global__ __launch_bounds__(64)
void attn_kernel(const __bf16* __restrict__ qkv, const __bf16* __restrict__ lite,
                 const float* __restrict__ rpb, const int* __restrict__ relidx,
                 const float* __restrict__ lam1_raw, const float* __restrict__ lam2_raw,
                 __bf16* __restrict__ xcat) {
  __shared__ float kS[kN * kHD];
  __shared__ float vS[kN * kHD];
  __shared__ float klS[kN * kHDL];
  __shared__ float vlS[kN * kHDL];
  __shared__ float biasS[kN * kN];
  __shared__ float pS[kN * kN];
  __shared__ float plS[kN * kN];

  const int bh = blockIdx.x;
  const int b = bh >> 5;
  const int h = bh & 31;
  const int tid = threadIdx.x;
  const size_t rowbase = (size_t)b * kN;

  for (int i = tid; i < kN * kHD; i += 64) {
    const int j = i >> 5, d = i & 31;
    const size_t base = (rowbase + j) * 3072 + (size_t)h * 32 + d;
    kS[i] = bf2f(qkv[base + 1024]);
    vS[i] = bf2f(qkv[base + 2048]);
  }
  for (int i = tid; i < kN * kHDL; i += 64) {
    const int j = i >> 3, d = i & 7;
    const size_t base = (rowbase + j) * 768 + (size_t)h * 8 + d;
    klS[i] = bf2f(lite[base + 256]);
    vlS[i] = bf2f(lite[base + 512]);
  }
  for (int i = tid; i < kN * kN; i += 64)
    biasS[i] = rpb[(size_t)relidx[i] * kH + h];
  __syncthreads();

  const int r = tid;
  if (r < kN) {
    float q[kHD], ql[kHDL];
    const size_t qb = (rowbase + r) * 3072 + (size_t)h * 32;
#pragma unroll
    for (int d = 0; d < kHD; ++d) q[d] = bf2f(qkv[qb + d]) * kScale;
    const size_t qlb = (rowbase + r) * 768 + (size_t)h * 8;
#pragma unroll
    for (int d = 0; d < kHDL; ++d) ql[d] = bf2f(lite[qlb + d]) * kScaleL;

    float mx = -3.0e38f;
    for (int j = 0; j < kN; ++j) {
      float s = biasS[r * kN + j];
#pragma unroll
      for (int d = 0; d < kHD; ++d) s += q[d] * kS[j * kHD + d];
      pS[r * kN + j] = s;
      mx = fmaxf(mx, s);
    }
    float sum = 0.0f;
    for (int j = 0; j < kN; ++j) {
      const float e = __expf(pS[r * kN + j] - mx);
      pS[r * kN + j] = e;
      sum += e;
    }
    const float inv = 1.0f / sum;

    float mxl = -3.0e38f;
    for (int j = 0; j < kN; ++j) {
      float s = biasS[r * kN + j];
#pragma unroll
      for (int d = 0; d < kHDL; ++d) s += ql[d] * klS[j * kHDL + d];
      plS[r * kN + j] = s;
      mxl = fmaxf(mxl, s);
    }
    float suml = 0.0f;
    for (int j = 0; j < kN; ++j) {
      const float e = __expf(plS[r * kN + j] - mxl);
      plS[r * kN + j] = e;
      suml += e;
    }
    const float invl = 1.0f / suml;

    const float lam1 = 1.0f / (1.0f + __expf(-lam1_raw[0]));
    const float lam2 = 1.0f / (1.0f + __expf(-lam2_raw[0]));
    for (int j = 0; j < kN; ++j) {
      const float p = pS[r * kN + j] * inv;
      pS[r * kN + j] = p;
      plS[r * kN + j] = lam1 * (plS[r * kN + j] * invl) - lam2 * p;
    }

    const size_t ob = (rowbase + r) * 1280;
#pragma unroll 4
    for (int d = 0; d < kHD; ++d) {
      float a = 0.0f;
      for (int j = 0; j < kN; ++j) a += pS[r * kN + j] * vS[j * kHD + d];
      xcat[ob + (size_t)h * 32 + d] = f2bf(a);
    }
#pragma unroll
    for (int d = 0; d < kHDL; ++d) {
      float a = 0.0f;
      for (int j = 0; j < kN; ++j) a += plS[r * kN + j] * vlS[j * kHDL + d];
      xcat[ob + 1024 + (size_t)h * 8 + d] = f2bf(a);
    }
  }
}

// ---------------------------------------------------------------------------
extern "C" void kernel_launch(void* const* d_in, const int* in_sizes, int n_in,
                              void* d_out, int out_size, void* d_ws, size_t ws_size,
                              hipStream_t stream) {
  (void)in_sizes; (void)n_in; (void)out_size; (void)ws_size;
  const float* x      = (const float*)d_in[0];
  const float* qkv_w  = (const float*)d_in[1];
  const float* qkv_b  = (const float*)d_in[2];
  const float* rpb    = (const float*)d_in[3];
  const int*   relidx = (const int*)d_in[4];
  const float* proj_w = (const float*)d_in[5];
  const float* proj_b = (const float*)d_in[6];
  const float* ql_w   = (const float*)d_in[7];
  const float* kl_w   = (const float*)d_in[8];
  const float* vl_w   = (const float*)d_in[9];
  const float* dp_w   = (const float*)d_in[10];
  const float* dp_b   = (const float*)d_in[11];
  const float* gamma  = (const float*)d_in[12];
  const float* lam1   = (const float*)d_in[13];
  const float* lam2   = (const float*)d_in[14];
  float* out = (float*)d_out;
  char* ws = (char*)d_ws;

  size_t off = 0;
  auto take = [&](size_t bytes) {
    size_t o = off;
    off += (bytes + 255) & ~(size_t)255;
    return o;
  };
  __bf16* xb    = (__bf16*)(ws + take((size_t)kRows * kC * 2));
  __bf16* wqkv  = (__bf16*)(ws + take((size_t)kC * 3072 * 2));
  __bf16* wlite = (__bf16*)(ws + take((size_t)kC * 768 * 2));
  __bf16* wcat  = (__bf16*)(ws + take((size_t)1280 * kC * 2));
  float*  bcat  = (float*)(ws + take((size_t)kC * 4));
  __bf16* qkvo  = (__bf16*)(ws + take((size_t)kRows * 3072 * 2));
  __bf16* liteo = (__bf16*)(ws + take((size_t)kRows * 768 * 2));
  __bf16* xcat  = (__bf16*)(ws + take((size_t)kRows * 1280 * 2));

  auto cvt = [&](const float* src, __bf16* dst, long long n, int cols, int ld, int coff) {
    cvt_sub<<<dim3((unsigned)((n + 255) / 256)), 256, 0, stream>>>(src, dst, n, cols, ld, coff);
  };
  cvt(x,     xb,    (long long)kRows * kC, kC,   kC,   0);
  cvt(qkv_w, wqkv,  (long long)kC * 3072,  3072, 3072, 0);
  cvt(ql_w,  wlite, (long long)kC * 256,   256,  768,  0);
  cvt(kl_w,  wlite, (long long)kC * 256,   256,  768,  256);
  cvt(vl_w,  wlite, (long long)kC * 256,   256,  768,  512);

  fuse_wcat<<<dim3(1280 * 1024 / 256), 256, 0, stream>>>(proj_w, dp_w, gamma, wcat);
  fuse_bias<<<dim3(4), 256, 0, stream>>>(proj_b, dp_b, proj_w, gamma, bcat);

  gemm_bf16_wmma<__bf16><<<dim3(3072 / 128, kRows / 128), 256, 0, stream>>>(
      xb, wqkv, qkv_b, qkvo, kRows, 3072, kC);
  gemm_bf16_wmma<__bf16><<<dim3(768 / 128, kRows / 128), 256, 0, stream>>>(
      xb, wlite, nullptr, liteo, kRows, 768, kC);
  attn_kernel<<<dim3(kB * kH), 64, 0, stream>>>(qkvo, liteo, rpb, relidx, lam1, lam2, xcat);
  gemm_bf16_wmma<float><<<dim3(1024 / 128, kRows / 128), 256, 0, stream>>>(
      xcat, wcat, bcat, out, kRows, 1024, 1280);
}
